// MVDreamI2VXFormersAttnProcessor_17214228922562
// MI455X (gfx1250) — matrix operand order repaired and checked by reading, not compiled, not run on hardware
//
#include <hip/hip_runtime.h>
#include <hip/hip_bf16.h>

typedef _Float16 h16;
typedef __attribute__((ext_vector_type(16))) _Float16 v16h;
typedef __attribute__((ext_vector_type(8)))  _Float16 v8h;
typedef __attribute__((ext_vector_type(8)))  float    v8f;
typedef __attribute__((ext_vector_type(4)))  float    v4f;

namespace {
constexpr int NV = 4;            // views
constexpr int NF = 8;            // frames
constexpr int LT = 256;          // tokens per (view,frame)
constexpr int C  = 320;          // channels
constexpr int NH = 8;            // heads
constexpr int BF = NF;           // b * f  (b == 1)
constexpr int L  = NV * LT;      // 1024 rearranged sequence length
constexpr int D  = C / NH;       // 40 head dim
constexpr int DP = 64;           // zero-padded head dim for Q/K
constexpr int DV = 48;           // zero-padded head-dim rows for V^T
constexpr int M  = BF * L;       // 8192 rows
}

#define WMMA(a, b, c) __builtin_amdgcn_wmma_f32_16x16x32_f16( \
    false, (a), false, (b), (short)0, (c), false, false)

__device__ __forceinline__ v16h cat8(v8h a, v8h b) {
  v16h r;
#pragma unroll
  for (int i = 0; i < 8; ++i) { r[i] = a[i]; r[i + 8] = b[i]; }
  return r;
}

// A-matrix (16x32 f16) fragment from row-major storage.
// Lanes 0-15: row M=lane, halves 0..7 -> K=0..7, halves 8..15 -> K=16..23.
// Lanes 16-31: row M=lane-16, K=8..15 and K=24..31.
__device__ __forceinline__ v16h load_a_f16(const h16* src, int ld) {
  const int lane = threadIdx.x & 31;
  const h16* p = src + (lane & 15) * ld + ((lane >> 4) & 1) * 8;
  return cat8(*(const v8h*)p, *(const v8h*)(p + 16));
}

// Same fragment but converting from f32 row-major on the fly.
__device__ __forceinline__ v16h load_a_f32cvt(const float* src, int ld) {
  const int lane = threadIdx.x & 31;
  const float* p = src + (lane & 15) * ld + ((lane >> 4) & 1) * 8;
  v4f f0 = *(const v4f*)(p);
  v4f f1 = *(const v4f*)(p + 4);
  v4f f2 = *(const v4f*)(p + 16);
  v4f f3 = *(const v4f*)(p + 20);
  v16h a;
#pragma unroll
  for (int i = 0; i < 4; ++i) {
    a[i]      = (h16)f0[i];
    a[4 + i]  = (h16)f1[i];
    a[8 + i]  = (h16)f2[i];
    a[12 + i] = (h16)f3[i];
  }
  return a;
}

// B-matrix (32x16 f16) fragment from *transposed* storage Bt[n][k]
// (i.e. B[k][n] = Bt[n][k]). Lane n (0-15): K=0..7,16..23, lane n+16: K=8..15,24..31.
__device__ __forceinline__ v16h load_bt(const h16* bt, int ld) {
  const int lane = threadIdx.x & 31;
  const h16* p = bt + (lane & 15) * ld + ((lane >> 4) & 1) * 8;
  return cat8(*(const v8h*)p, *(const v8h*)(p + 16));
}

// ---------------------------------------------------------------- pack W^T
__global__ void pack_w_kernel(const float* __restrict__ w, h16* __restrict__ wt) {
  int idx = blockIdx.x * blockDim.x + threadIdx.x;
  if (idx < C * C) {
    int n = idx / C, k = idx % C;
    wt[idx] = (h16)w[k * C + n];       // Wt[n][k] = W[k][n]
  }
}

// --------------------------------------------------- fused QKV(+Qi2v) proj
// grid: (M/16, C/64, 4), block: 32. One wave = 16x64 tile, A-frag reused 4x.
__global__ __launch_bounds__(32) void proj_kernel(
    const float* __restrict__ hs,    // [NV*NF*LT][C]  (b n f) l c layout
    const h16* __restrict__ wt4,     // 4 x [C][C] transposed f16: q,k,v,qi
    h16* __restrict__ qb,            // [BF][NH][L][DP]  zero pre-padded
    h16* __restrict__ kb,            // [BF][NH][L][DP]
    h16* __restrict__ vt,            // [BF][NH][DV][L]
    h16* __restrict__ qib)           // [BF][NH][L][DP]
{
  const int mt = blockIdx.x, nt = blockIdx.y, which = blockIdx.z;
  const int m0 = mt * 16, n0 = nt * 64;
  // h row -> hidden_states row (rearrange folded in); contiguous inside tile
  const int bf  = m0 / L;
  const int pos = m0 % L;
  const int nv  = pos / LT;
  const int jj  = pos % LT;
  const int srow = (nv * NF + bf) * LT + jj;

  const float* A = hs + (size_t)srow * C;
  const h16*   B = wt4 + (size_t)which * C * C + (size_t)n0 * C;

  v8f acc[4] = {};
#pragma unroll
  for (int k = 0; k < C; k += 32) {
    __builtin_prefetch((const void*)(A + k + 96), 0, 0);
    v16h a = load_a_f32cvt(A + k, C);
#pragma unroll
    for (int t = 0; t < 4; ++t) {
      v16h b = load_bt(B + (size_t)t * 16 * C + k, C);
      acc[t] = WMMA(a, b, acc[t]);
    }
  }

  const int lane = threadIdx.x & 31;
  const int l = lane & 15, hi = (lane >> 4) & 1;
#pragma unroll
  for (int t = 0; t < 4; ++t) {
#pragma unroll
    for (int r = 0; r < 8; ++r) {
      int row = m0 + 8 * hi + r;          // h-space row
      int c   = n0 + t * 16 + l;          // 0..319
      int head = c / D, d = c % D;
      int bfr = row / L, posr = row % L;
      size_t bh = (size_t)(bfr * NH + head);
      h16 val = (h16)acc[t][r];
      if (which == 0)      qb [(bh * L + posr) * DP + d] = val;
      else if (which == 1) kb [(bh * L + posr) * DP + d] = val;
      else if (which == 2) vt [(bh * DV + d) * L + posr] = val;   // transposed
      else                 qib[(bh * L + posr) * DP + d] = val;
    }
  }
}

// ----------------------------------------------------------- attention
// grid: (L/32, BF*NH, 2), block: 32. One wave = 32 query rows (2 row groups),
// K/V fragments reused across both groups; softmax: one row per lane.
__global__ __launch_bounds__(32) void attn_kernel(
    const h16* __restrict__ qb, const h16* __restrict__ kb,
    const h16* __restrict__ vt, const h16* __restrict__ qib,
    h16* __restrict__ obase,     // [BF][L][C]  (head-concat)
    h16* __restrict__ oi2v)
{
  const int lane  = threadIdx.x & 31;
  const int qt    = blockIdx.x;                   // 32-row query tile
  const int bh    = blockIdx.y;
  const int which = blockIdx.z;
  const int bf = bh / NH, hh = bh % NH;
  const int kvbf = which ? (bf / NF) * NF : bf;   // first frame for i2v

  const h16* Q  = (which ? qib : qb) + ((size_t)bh * L + (size_t)qt * 32) * DP;
  const h16* K  = kb + (size_t)(kvbf * NH + hh) * L * DP;
  const h16* VT = vt + (size_t)(kvbf * NH + hh) * DV * L;
  h16* OUT = (which ? oi2v : obase) + ((size_t)bf * L + (size_t)qt * 32) * C + hh * D;

  __shared__ __align__(16) float sS[32][33];
  __shared__ __align__(16) h16   sP[32][32];
  __shared__ float sCorr[32];
  __shared__ float sInv[32];

  const int l = lane & 15, hi = (lane >> 4) & 1;
  const float scale = 0.15811388300841897f;       // 1/sqrt(40)

  v16h qa[2][2];
#pragma unroll
  for (int g = 0; g < 2; ++g) {
    qa[g][0] = load_a_f16(Q + (size_t)g * 16 * DP, DP);       // d = 0..31
    qa[g][1] = load_a_f16(Q + (size_t)g * 16 * DP + 32, DP);  // d = 32..63 (pad)
  }
  v8f o[2][3] = {};
  float m_run = -1e30f, l_run = 0.0f;             // one query row per lane

  for (int j = 0; j < L; j += 32) {
    __builtin_prefetch((const void*)(K + (size_t)(j + 32) * DP), 0, 0);
    // ---- K fragments for this key chunk, shared by both row groups
    v16h kb00 = load_bt(K + (size_t)j * DP, DP);
    v16h kb01 = load_bt(K + (size_t)j * DP + 32, DP);
    v16h kb10 = load_bt(K + (size_t)(j + 16) * DP, DP);
    v16h kb11 = load_bt(K + (size_t)(j + 16) * DP + 32, DP);
#pragma unroll
    for (int g = 0; g < 2; ++g) {
      v8f s0 = {}, s1 = {};
      s0 = WMMA(qa[g][0], kb00, s0); s0 = WMMA(qa[g][1], kb01, s0);
      s1 = WMMA(qa[g][0], kb10, s1); s1 = WMMA(qa[g][1], kb11, s1);
#pragma unroll
      for (int r = 0; r < 8; ++r) {
        sS[g * 16 + 8 * hi + r][l]      = s0[r] * scale;
        sS[g * 16 + 8 * hi + r][16 + l] = s1[r] * scale;
      }
    }
    __syncthreads();

    // ---- online softmax: lane == query row (all 32 lanes active)
    {
      float cm = -1e30f;
#pragma unroll
      for (int c = 0; c < 32; ++c) cm = fmaxf(cm, sS[lane][c]);
      float nm   = fmaxf(m_run, cm);
      float corr = __expf(m_run - nm);
      float s = 0.0f;
#pragma unroll
      for (int c = 0; c < 32; ++c) {
        float e = __expf(sS[lane][c] - nm);
        sP[lane][c] = (h16)e;
        s += e;
      }
      l_run = l_run * corr + s;
      m_run = nm;
      sCorr[lane] = corr;
    }
    __syncthreads();

    // ---- rescale O, accumulate P(32x32) @ V(32x48); V frags shared
#pragma unroll
    for (int g = 0; g < 2; ++g)
#pragma unroll
      for (int r = 0; r < 8; ++r) {
        float cr = sCorr[g * 16 + 8 * hi + r];
        o[g][0][r] *= cr; o[g][1][r] *= cr; o[g][2][r] *= cr;
      }
    v16h vb0 = load_bt(VT + (size_t)0  * L + j, L);
    v16h vb1 = load_bt(VT + (size_t)16 * L + j, L);
    v16h vb2 = load_bt(VT + (size_t)32 * L + j, L);
#pragma unroll
    for (int g = 0; g < 2; ++g) {
      v16h pa = load_a_f16(&sP[g * 16][0], 32);   // ds_load_b128 x2
      o[g][0] = WMMA(pa, vb0, o[g][0]);
      o[g][1] = WMMA(pa, vb1, o[g][1]);
      o[g][2] = WMMA(pa, vb2, o[g][2]);
    }
    __syncthreads();
  }

  sInv[lane] = 1.0f / l_run;
  __syncthreads();

#pragma unroll
  for (int g = 0; g < 2; ++g)
#pragma unroll
    for (int r = 0; r < 8; ++r) {
      int row = g * 16 + 8 * hi + r;
      float inv = sInv[row];
      OUT[row * C + l]      = (h16)(o[g][0][r] * inv);
      OUT[row * C + 16 + l] = (h16)(o[g][1][r] * inv);
      if (32 + l < D) OUT[row * C + 32 + l] = (h16)(o[g][2][r] * inv);
    }
}

// ----------------------------- t = add + A @ Wt + bias   (f16 out)
// grid: (M/16, C/64), block: 32. 16x64 per wave.
__global__ __launch_bounds__(32) void gemm_add_kernel(
    const h16* __restrict__ Abuf,   // [M][C] f16
    const h16* __restrict__ wt,     // [C][C] transposed f16
    const float* __restrict__ bias,
    const h16* __restrict__ addbuf, // [M][C] f16
    h16* __restrict__ outbuf)       // [M][C] f16
{
  const int m0 = blockIdx.x * 16, n0 = blockIdx.y * 64;
  const h16* A = Abuf + (size_t)m0 * C;
  const h16* B = wt + (size_t)n0 * C;
  v8f acc[4] = {};
#pragma unroll
  for (int k = 0; k < C; k += 32) {
    __builtin_prefetch((const void*)(A + k + 96), 0, 0);
    v16h a = load_a_f16(A + k, C);
#pragma unroll
    for (int t = 0; t < 4; ++t)
      acc[t] = WMMA(a, load_bt(B + (size_t)t * 16 * C + k, C), acc[t]);
  }

  const int lane = threadIdx.x & 31;
  const int l = lane & 15, hi = (lane >> 4) & 1;
#pragma unroll
  for (int t = 0; t < 4; ++t)
#pragma unroll
    for (int r = 0; r < 8; ++r) {
      int row = m0 + 8 * hi + r, c = n0 + t * 16 + l;
      outbuf[(size_t)row * C + c] =
          (h16)(acc[t][r] + bias[c] + (float)addbuf[(size_t)row * C + c]);
    }
}

// ----------------------------- out = A @ Wt + bias, inverse rearrange, f32
__global__ __launch_bounds__(32) void gemm_out_kernel(
    const h16* __restrict__ Abuf,   // [M][C] f16 in (b f)(n l) order
    const h16* __restrict__ wt,     // [C][C] transposed f16
    const float* __restrict__ bias,
    float* __restrict__ out)        // [(b n f) l][C] f32
{
  const int m0 = blockIdx.x * 16, n0 = blockIdx.y * 64;
  const h16* A = Abuf + (size_t)m0 * C;
  const h16* B = wt + (size_t)n0 * C;
  v8f acc[4] = {};
#pragma unroll
  for (int k = 0; k < C; k += 32) {
    __builtin_prefetch((const void*)(A + k + 96), 0, 0);
    v16h a = load_a_f16(A + k, C);
#pragma unroll
    for (int t = 0; t < 4; ++t)
      acc[t] = WMMA(a, load_bt(B + (size_t)t * 16 * C + k, C), acc[t]);
  }

  const int lane = threadIdx.x & 31;
  const int l = lane & 15, hi = (lane >> 4) & 1;
#pragma unroll
  for (int t = 0; t < 4; ++t)
#pragma unroll
    for (int r = 0; r < 8; ++r) {
      int row = m0 + 8 * hi + r;          // h-space row
      int c = n0 + t * 16 + l;
      int bf = row / L, pos = row % L;
      int nv = pos / LT, jj = pos % LT;
      int drow = (nv * NF + bf) * LT + jj;
      out[(size_t)drow * C + c] = acc[t][r] + bias[c];
    }
}

extern "C" void kernel_launch(void* const* d_in, const int* in_sizes, int n_in,
                              void* d_out, int out_size, void* d_ws, size_t ws_size,
                              hipStream_t stream) {
  (void)in_sizes; (void)n_in; (void)out_size; (void)ws_size;
  const float* hs      = (const float*)d_in[0];
  const float* Wq      = (const float*)d_in[1];
  const float* Wk      = (const float*)d_in[2];
  const float* Wv      = (const float*)d_in[3];
  const float* Wo      = (const float*)d_in[4];
  const float* bo      = (const float*)d_in[5];
  const float* Wq_i2v  = (const float*)d_in[6];
  const float* Wo_i2v  = (const float*)d_in[7];
  const float* bo_i2v  = (const float*)d_in[8];

  // workspace carve-up (all region sizes are multiples of 16 bytes)
  h16* wt6  = (h16*)d_ws;                         // 6 x [C][C] f16 W^T
  h16* qb   = wt6 + (size_t)6 * C * C;            // [BF][NH][L][DP]
  h16* kbuf = qb   + (size_t)BF * NH * L * DP;
  h16* qib  = kbuf + (size_t)BF * NH * L * DP;
  h16* vtb  = qib  + (size_t)BF * NH * L * DP;    // [BF][NH][DV][L]
  h16* baseb = vtb + (size_t)BF * NH * DV * L;    // [BF][L][C]
  h16* i2vb  = baseb + (size_t)BF * L * C;
  h16* tbuf  = i2vb  + (size_t)BF * L * C;

  // 1) pack weights f32 -> f16 transposed. order: q,k,v,qi | oi2v | o
  dim3 pw((C * C + 255) / 256);
  pack_w_kernel<<<pw, 256, 0, stream>>>(Wq,     wt6 + (size_t)0 * C * C);
  pack_w_kernel<<<pw, 256, 0, stream>>>(Wk,     wt6 + (size_t)1 * C * C);
  pack_w_kernel<<<pw, 256, 0, stream>>>(Wv,     wt6 + (size_t)2 * C * C);
  pack_w_kernel<<<pw, 256, 0, stream>>>(Wq_i2v, wt6 + (size_t)3 * C * C);
  pack_w_kernel<<<pw, 256, 0, stream>>>(Wo_i2v, wt6 + (size_t)4 * C * C);
  pack_w_kernel<<<pw, 256, 0, stream>>>(Wo,     wt6 + (size_t)5 * C * C);

  // 2) zero the padded q/k/qi/vt region (contiguous) so WMMA pads are 0
  size_t zbytes = ((size_t)3 * BF * NH * L * DP + (size_t)BF * NH * DV * L) * sizeof(h16);
  hipMemsetAsync(qb, 0, zbytes, stream);

  // 3) projections (rearrange folded into row index)
  proj_kernel<<<dim3(M / 16, C / 64, 4), 32, 0, stream>>>(hs, wt6, qb, kbuf, vtb, qib);

  // 4) both attentions (z=0 self, z=1 first-frame KV)
  attn_kernel<<<dim3(L / 32, BF * NH, 2), 32, 0, stream>>>(qb, kbuf, vtb, qib, baseb, i2vb);

  // 5) t = base + i2v @ Wo_i2v + bo_i2v
  gemm_add_kernel<<<dim3(M / 16, C / 64), 32, 0, stream>>>(
      i2vb, wt6 + (size_t)4 * C * C, bo_i2v, baseb, tbuf);

  // 6) out = t @ Wo + bo with inverse rearrange, f32 store
  gemm_out_kernel<<<dim3(M / 16, C / 64), 32, 0, stream>>>(
      tbuf, wt6 + (size_t)5 * C * C, bo, (float*)d_out);
}
